// TopicAwareModel_31430570672673
// MI455X (gfx1250) — matrix operand order repaired and checked by reading, not compiled, not run on hardware
//
#include <hip/hip_runtime.h>
#include <hip/hip_bf16.h>
#include <math.h>

// ---- problem constants (match reference) ----
#define Bz   8
#define Sz   20
#define Fz   200
#define Dz   2048
#define Tz   20
#define Ez   128
#define Hz   128
#define THz  128
#define CDz  300
#define Mz   (Bz * Sz * Fz)      // 32000 flattened rows
#define RPB  (Sz * Fz)           // rows per batch element = 4000
#define NKSTEP (Dz / 32)         // 64 K-chunks of 32

typedef __bf16 bf16_t;
typedef bf16_t v16bf __attribute__((ext_vector_type(16)));
typedef float  v8f   __attribute__((ext_vector_type(8)));

// ---- workspace layout (floats) ----
#define WS_POOLED   0                        // [Bz*Dz]   raw (undivided) pooled sums
#define WS_VW3      (WS_POOLED + Bz * Dz)    // [Bz*Hz]   video @ W3
#define WS_TPROB    (WS_VW3 + Bz * Hz)       // [Bz*Tz]   softmax topic probs
#define WS_TC       (WS_TPROB + Bz * Tz)     // [Tz*Hz]   topic_emb @ W2
#define WS_W1PACK   (WS_TC + Tz * Hz)        // [Dz*Hz] bf16 (=Dz*Hz/2 floats), fragment-swizzled

__device__ __forceinline__ bf16_t f2bf(float x) { return (bf16_t)x; }

// CDNA5 async LDS-DMA: copies 16B per lane, mem -> LDS, tracked by ASYNCcnt.
// INST_OFFSET applies to both global and LDS addresses (ISA §10 async pseudocode).
__device__ __forceinline__ void async_b128(unsigned int lds_addr, const void* gaddr) {
    asm volatile("global_load_async_to_lds_b128 %0, %1, off"
                 :: "v"(lds_addr), "v"(gaddr) : "memory");
}
__device__ __forceinline__ void async_b128_off16(unsigned int lds_addr, const void* gaddr) {
    asm volatile("global_load_async_to_lds_b128 %0, %1, off offset:16"
                 :: "v"(lds_addr), "v"(gaddr) : "memory");
}
__device__ __forceinline__ void wait_asynccnt0() {
    asm volatile("s_wait_asynccnt 0x0" ::: "memory");
}

// =====================================================================
// Kernel 1: masked sum-pool over (s,f).  grid (Dz/1024, Bz), block 256.
// =====================================================================
__global__ __launch_bounds__(256) void pool_kernel(const float* __restrict__ batch,
                                                   const int* __restrict__ seg_len,
                                                   float* __restrict__ ws) {
    const int b  = blockIdx.y;
    const int k0 = (blockIdx.x * blockDim.x + threadIdx.x) * 4;
    if (k0 >= Dz) return;
    float4 acc = make_float4(0.f, 0.f, 0.f, 0.f);
    for (int s = 0; s < Sz; ++s) {
        const int len = seg_len[b * Sz + s];
        const float* base = batch + ((size_t)(b * Sz + s) * Fz) * Dz + k0;
        for (int f = 0; f < len; ++f) {
            const float4 v = *(const float4*)(base + (size_t)f * Dz);
            acc.x += v.x; acc.y += v.y; acc.z += v.z; acc.w += v.w;
        }
    }
    float* dst = ws + WS_POOLED + b * Dz + k0;
    dst[0] = acc.x; dst[1] = acc.y; dst[2] = acc.z; dst[3] = acc.w;
}

// =====================================================================
// Kernel 1b: convert W1 -> bf16, pre-swizzled into WMMA B-fragment order.
// Chunk c (K rows c*32..c*32+31): linear layout [ntile][lane][slot16].
// grid 64 blocks x 256 threads; block = one K-chunk.
// =====================================================================
__global__ __launch_bounds__(256) void pack_w1_kernel(const float* __restrict__ W1,
                                                      float* __restrict__ ws) {
    bf16_t* wp = (bf16_t*)(ws + WS_W1PACK);
    const int tid  = threadIdx.x;
    const int c    = blockIdx.x;
    const int kloc = tid >> 3;        // 0..31  (K within chunk)
    const int nt   = tid & 7;         // ntile
    const int lhalf = kloc & 16;      // K-half -> lane group
    const int slot  = kloc & 15;
    const float* src = W1 + (size_t)(c * 32 + kloc) * Hz + nt * 16;
    bf16_t* dst = wp + (size_t)c * 4096 + nt * 512 + slot;
    #pragma unroll
    for (int i = 0; i < 16; ++i)
        dst[(lhalf + i) * 16] = f2bf(src[i]);
}

// =====================================================================
// Kernel 2: tiny "head" — video, topic probs, vW3, tc.  <<<1, 256>>>
// =====================================================================
__global__ __launch_bounds__(256) void head_kernel(
                            const int*   __restrict__ seg_len,
                            const float* __restrict__ concept1,
                            const float* __restrict__ concept2,
                            const float* __restrict__ W_enc,
                            const float* __restrict__ b_enc,
                            const float* __restrict__ Wc1,
                            const float* __restrict__ Wc2,
                            const float* __restrict__ Wt1,
                            const float* __restrict__ bt1,
                            const float* __restrict__ Wt2,
                            const float* __restrict__ bt2,
                            const float* __restrict__ topic_emb,
                            const float* __restrict__ W2,
                            const float* __restrict__ W3,
                            float* __restrict__ ws) {
    __shared__ float s_video[Bz][Hz];
    __shared__ float s_th[Bz][THz];
    __shared__ float s_logit[Bz][Tz];
    __shared__ float s_denom[Bz];

    const int tid = threadIdx.x;

    if (tid < Bz) {
        int d = 0;
        for (int s = 0; s < Sz; ++s) d += seg_len[tid * Sz + s];
        s_denom[tid] = fmaxf((float)d, 1.0f);
    }
    __syncthreads();

    const float* pooled = ws + WS_POOLED;
    for (int q = 0; q < (Bz * Hz) / 256; ++q) {
        const int o = tid + 256 * q;
        const int b = o >> 7, h = o & (Hz - 1);
        float ap = 0.f;
        for (int k = 0; k < Dz; ++k) ap = fmaf(pooled[b * Dz + k], W_enc[k * Hz + h], ap);
        float ac = 0.f;
        for (int k = 0; k < CDz; ++k) {
            ac = fmaf(concept1[b * CDz + k], Wc1[k * Hz + h], ac);
            ac = fmaf(concept2[b * CDz + k], Wc2[k * Hz + h], ac);
        }
        s_video[b][h] = fmaxf(ap / s_denom[b] + b_enc[h] + ac, 0.f);
    }
    __syncthreads();

    for (int q = 0; q < (Bz * THz) / 256; ++q) {
        const int o = tid + 256 * q;
        const int b = o >> 7, h = o & (THz - 1);
        float a = bt1[h];
        for (int k = 0; k < Hz; ++k) a = fmaf(s_video[b][k], Wt1[k * THz + h], a);
        s_th[b][h] = fmaxf(a, 0.f);
    }
    __syncthreads();

    if (tid < Bz * Tz) {
        const int b = tid / Tz, t = tid % Tz;
        float a = bt2[t];
        for (int k = 0; k < THz; ++k) a = fmaf(s_th[b][k], Wt2[k * Tz + t], a);
        s_logit[b][t] = a;
    }
    __syncthreads();

    if (tid < Bz) {
        float m = s_logit[tid][0];
        for (int t = 1; t < Tz; ++t) m = fmaxf(m, s_logit[tid][t]);
        float sum = 0.f;
        float e[Tz];
        for (int t = 0; t < Tz; ++t) { e[t] = expf(s_logit[tid][t] - m); sum += e[t]; }
        for (int t = 0; t < Tz; ++t) ws[WS_TPROB + tid * Tz + t] = e[t] / sum;
    }

    for (int q = 0; q < (Bz * Hz) / 256; ++q) {
        const int o = tid + 256 * q;
        const int b = o >> 7, h = o & (Hz - 1);
        float a = 0.f;
        for (int k = 0; k < Hz; ++k) a = fmaf(s_video[b][k], W3[k * Hz + h], a);
        ws[WS_VW3 + b * Hz + h] = a;
    }

    for (int q = 0; q < (Tz * Hz) / 256; ++q) {
        const int o = tid + 256 * q;
        const int t = o >> 7, h = o & (Hz - 1);
        float a = 0.f;
        for (int e2 = 0; e2 < Ez; ++e2) a = fmaf(topic_emb[t * Ez + e2], W2[e2 * Hz + h], a);
        ws[WS_TC + t * Hz + h] = a;
    }
}

// =====================================================================
// Kernel 3: base = batch @ W1 (bf16 WMMA, f32 acc), fused topic loop.
// Pre-swizzled bf16 W1 chunks are streamed into double-buffered LDS via
// async LDS-DMA (ASYNCcnt) while WMMAs consume the other buffer.
// Block = 256 threads (8 waves). Wave owns 16 rows x 128 cols. grid = 250.
// __launch_bounds__(256, 1): single workgroup per WGP -> full VGPR budget,
// no scratch spills in the pipelined main loop.
// =====================================================================
__global__ __launch_bounds__(256, 1) void gemm_topic_kernel(
                                  const float* __restrict__ batch,   // [Mz, Dz]
                                  const float* __restrict__ b1,      // [Hz]
                                  const float* __restrict__ w_out,   // [Hz]
                                  const float* __restrict__ b_out_p, // scalar
                                  const float* __restrict__ ws,      // vW3/tprob/tc/W1pack
                                  float* __restrict__ out) {         // [2 * Mz]
    // double-buffered B fragments: [buf][ntile][lane][slot(16 bf16)] = 16 KB
    __shared__ __align__(32) bf16_t ldsB[2][8][32][16];

    const int tid  = threadIdx.x;
    const int wave = tid >> 5;
    const int lane = tid & 31;
    const int l15  = lane & 15;
    const int half = lane >> 4;
    const int rowTile = blockIdx.x * 128 + wave * 16;

    const float*  vW3 = ws + WS_VW3;
    const float*  tpr = ws + WS_TPROB;
    const float*  tc  = ws + WS_TC;
    const bf16_t* wp  = (const bf16_t*)(ws + WS_W1PACK);

    v8f acc[8];
    #pragma unroll
    for (int j = 0; j < 8; ++j) acc[j] = (v8f)0.f;

    const float* arow = batch + (size_t)(rowTile + l15) * Dz;

    // this thread's 32-byte slice of each 8 KB chunk (layout matches LDS)
    const unsigned int ldsBase0 =
        (unsigned int)(uintptr_t)(&ldsB[0][0][0][0]) + (unsigned int)tid * 32u;
    const unsigned int ldsBase1 =
        (unsigned int)(uintptr_t)(&ldsB[1][0][0][0]) + (unsigned int)tid * 32u;

    // prologue: stage chunk 0 into buffer 0
    {
        const bf16_t* g = wp + (size_t)tid * 16;
        async_b128(ldsBase0, g);
        async_b128_off16(ldsBase0, g);
    }

    for (int step = 0; step < NKSTEP; ++step) {
        const int p  = step & 1;
        const int kk = step * 32;

        wait_asynccnt0();      // this thread's chunk `step` landed in LDS
        __syncthreads();       // everyone's landed; everyone done reading buf 1-p

        if (step + 1 < NKSTEP) {   // stage next chunk into the other buffer
            const bf16_t* g = wp + (size_t)(step + 1) * 4096 + (size_t)tid * 16;
            const unsigned int dst = p ? ldsBase0 : ldsBase1;
            async_b128(dst, g);
            async_b128_off16(dst, g);
        }

        // --- A fragment, contiguous K order k = 16*half + slot, same as packed B (fp32 -> bf16 inline) ---
        v16bf afrag;
        {
            const int c0 = kk + half * 16;       // V0-3: K 0-7 / 16-23
            const int c1 = kk + half * 16 + 8;   // V4-7: K 8-15 / 24-31
            #pragma unroll
            for (int i = 0; i < 8; ++i) afrag[i]     = f2bf(arow[c0 + i]);
            #pragma unroll
            for (int i = 0; i < 8; ++i) afrag[8 + i] = f2bf(arow[c1 + i]);
        }
        if (kk + 32 < Dz) __builtin_prefetch(arow + kk + 32, 0, 1);

        // --- 8 WMMAs: A (16x32 bf16) x B (32x16 bf16) += C (16x16 f32) ---
        #pragma unroll
        for (int j = 0; j < 8; ++j) {
            v16bf bfrag = *(const v16bf*)&ldsB[p][j][lane][0];
            acc[j] = __builtin_amdgcn_wmma_f32_16x16x32_bf16(
                false, afrag, false, bfrag, (short)0, acc[j], false, false);
        }
    }

    // ================= epilogue (all in registers) =================
    const float b_out = b_out_p[0];
    float b1n[8], wj[8];
    #pragma unroll
    for (int j = 0; j < 8; ++j) {
        b1n[j] = b1[j * 16 + l15];
        wj[j]  = w_out[j * 16 + l15];
    }

    float basev[8][8];
    int   bidx[8];
    #pragma unroll
    for (int i = 0; i < 8; ++i) {
        const int row = rowTile + half * 8 + i;
        const int b   = row / RPB;
        bidx[i] = b;
        #pragma unroll
        for (int j = 0; j < 8; ++j)
            basev[i][j] = acc[j][i] + b1n[j] + vW3[b * Hz + j * 16 + l15];
    }

    float ov[8];
    #pragma unroll
    for (int i = 0; i < 8; ++i) ov[i] = 0.f;

    for (int t = 0; t < Tz; ++t) {
        float tcv[8];
        #pragma unroll
        for (int j = 0; j < 8; ++j) tcv[j] = tc[t * Hz + j * 16 + l15];
        #pragma unroll
        for (int i = 0; i < 8; ++i) {
            float s = 0.f;
            #pragma unroll
            for (int j = 0; j < 8; ++j)
                s = fmaf(fmaxf(basev[i][j] + tcv[j], 0.f), wj[j], s);
            s += __shfl_xor(s, 1);
            s += __shfl_xor(s, 2);
            s += __shfl_xor(s, 4);
            s += __shfl_xor(s, 8);
            const float sig = 1.f / (1.f + expf(-(s + b_out)));
            const float sc  = tpr[bidx[i] * Tz + t] * sig - 0.01f;
            ov[i] += fmaxf(sc, 0.f);
        }
    }

    if (l15 == 0) {
        #pragma unroll
        for (int i = 0; i < 8; ++i) {
            const int row = rowTile + half * 8 + i;
            const float v = ov[i] * (1.f / (float)Tz);
            out[row]      = v;
            out[Mz + row] = v;
        }
    }
}

// =====================================================================
extern "C" void kernel_launch(void* const* d_in, const int* in_sizes, int n_in,
                              void* d_out, int out_size, void* d_ws, size_t ws_size,
                              hipStream_t stream) {
    const float* batch     = (const float*)d_in[0];
    const int*   seg_len   = (const int*)  d_in[1];
    const float* concept1  = (const float*)d_in[2];
    const float* concept2  = (const float*)d_in[3];
    const float* W_enc     = (const float*)d_in[4];
    const float* b_enc     = (const float*)d_in[5];
    const float* Wc1       = (const float*)d_in[6];
    const float* Wc2       = (const float*)d_in[7];
    const float* Wt1       = (const float*)d_in[8];
    const float* bt1       = (const float*)d_in[9];
    const float* Wt2       = (const float*)d_in[10];
    const float* bt2       = (const float*)d_in[11];
    const float* topic_emb = (const float*)d_in[12];
    const float* W1        = (const float*)d_in[13];
    const float* b1        = (const float*)d_in[14];
    const float* W2        = (const float*)d_in[15];
    const float* W3        = (const float*)d_in[16];
    const float* w_out     = (const float*)d_in[17];
    const float* b_out     = (const float*)d_in[18];

    float* ws  = (float*)d_ws;
    float* out = (float*)d_out;

    // 1) masked pool (raw sums)
    dim3 g1(Dz / (256 * 4), Bz);
    pool_kernel<<<g1, 256, 0, stream>>>(batch, seg_len, ws);

    // 1b) W1 -> bf16 fragment-swizzled pack (512 KB, L2-resident)
    pack_w1_kernel<<<NKSTEP, 256, 0, stream>>>(W1, ws);

    // 2) head: video / topic probs / vW3 / tc
    head_kernel<<<1, 256, 0, stream>>>(seg_len, concept1, concept2,
                                       W_enc, b_enc, Wc1, Wc2,
                                       Wt1, bt1, Wt2, bt2,
                                       topic_emb, W2, W3, ws);

    // 3) WMMA GEMM + fused topic loop (async LDS-DMA double buffering)
    gemm_topic_kernel<<<Mz / 128, 256, 0, stream>>>(batch, b1, w_out, b_out,
                                                    ws, out);
}
